// AttentionRNNCell_16338055594139
// MI455X (gfx1250) — matrix operand order repaired and checked by reading, compiled, not run on hardware
//
#include <hip/hip_runtime.h>
#include <hip/hip_bf16.h>

typedef __attribute__((ext_vector_type(16))) _Float16 v16h;
typedef __attribute__((ext_vector_type(8)))  _Float16 v8h;
typedef __attribute__((ext_vector_type(8)))  float    v8f;
typedef __attribute__((ext_vector_type(4)))  float    v4f;

#define BB 512
#define TT 256
#define HH 512
#define DD 1024      // H + E (encoder feature dim)
#define GG 1536      // 2H + E = 3H
#define CC 5000
#define KPAD 5024    // C padded to multiple of 32 (K of embed GEMM)
#define CPAD 5008    // C padded to multiple of 16 (N of out GEMM)

// ---------------------------------------------------------------------------
// Generic WMMA f16 GEMM:  C[m,n] = sum_k A[m,k] * B[n,k] + bias[n]
// One wave per 16x16 output tile. M multiple of 16, K multiple of 32.
// N handled with row-clamped B loads; ldc must be >= gridDim.x*16.
// ---------------------------------------------------------------------------
__global__ void wmma_gemm_kernel(const _Float16* __restrict__ A, int lda,
                                 const _Float16* __restrict__ Bm, int ldb, int Nrows,
                                 const float* __restrict__ bias,
                                 float* __restrict__ Cout, int ldc, int K)
{
    const int lane  = threadIdx.x & 31;
    const int hi    = lane >> 4;        // 0: lanes 0-15, 1: lanes 16-31
    const int col16 = lane & 15;
    const int n0 = blockIdx.x * 16;
    const int m0 = blockIdx.y * 16;

    const int arow = m0 + col16;
    int brow = n0 + col16; if (brow >= Nrows) brow = Nrows - 1;

    const _Float16* Ar = A  + (size_t)arow * lda;
    const _Float16* Br = Bm + (size_t)brow * ldb;

    v8f acc = {};
    for (int k0 = 0; k0 < K; k0 += 32) {
        // A fragment (16x32): lane holds row (lane&15), 16 contiguous K at k0 + hi*16
        union { v16h v; v8h h[2]; } a;
        a.v = *(const v16h*)(Ar + k0 + hi * 16);
        // B fragment (32x16): lane holds col (lane&15), 16 contiguous K at k0 + hi*16
        v16h bf = *(const v16h*)(Br + k0 + hi * 16);
        acc = __builtin_amdgcn_wmma_f32_16x16x32_f16(false, a.v, false, bf,
                                                     (short)0, acc, false, false);
    }

    const int col = n0 + col16;
    int bcol = col < Nrows ? col : Nrows - 1;
    const float bv = bias[bcol];
#pragma unroll
    for (int i = 0; i < 8; ++i) {
        const int m = m0 + i + hi * 8;
        Cout[(size_t)m * ldc + col] = acc[i] + bv;
    }
}

// ---------------------------------------------------------------------------
// Fused attention-score kernel:
// scores[b,t] = sum_h v[h] * tanh( preh[b,h] + sum_k enc[t,b,k]*attnW[h,H+k] )
// WG = 128 threads (4 waves), handles one b and 64 consecutive t.
// enc tile (64 x 1024) staged once in LDS as f16; reused across all 32 N-tiles.
// ---------------------------------------------------------------------------
__global__ void attn_scores_kernel(const float* __restrict__ enc,      // (T,B,D)
                                   const _Float16* __restrict__ attnW16, // (H, G) f16
                                   const float* __restrict__ preh,     // (B,H) incl attn_b
                                   const float* __restrict__ vvec,     // (H)
                                   float* __restrict__ scores)         // (B,T)
{
    extern __shared__ _Float16 smem[];      // 64 * DD halves = 128 KB
    const int b  = blockIdx.y;
    const int t0 = blockIdx.x * 64;
    const int tid = threadIdx.x;            // 0..127

    // Stage 64 rows of enc (f32 -> f16), coalesced.
    for (int r = 0; r < 64; ++r) {
        const float* src = enc + ((size_t)(t0 + r) * BB + b) * DD;
        for (int k = tid * 4; k < DD; k += 128 * 4) {
            v4f f = *(const v4f*)(src + k);
            smem[r * DD + k + 0] = (_Float16)f[0];
            smem[r * DD + k + 1] = (_Float16)f[1];
            smem[r * DD + k + 2] = (_Float16)f[2];
            smem[r * DD + k + 3] = (_Float16)f[3];
        }
    }
    __syncthreads();

    const int w     = tid >> 5;
    const int lane  = tid & 31;
    const int hi    = lane >> 4;
    const int col16 = lane & 15;

    const _Float16* Arow = smem + (size_t)(w * 16 + col16) * DD;

    float sacc[8] = {0.f, 0.f, 0.f, 0.f, 0.f, 0.f, 0.f, 0.f};

    for (int n0 = 0; n0 < HH; n0 += 16) {
        const _Float16* Br = attnW16 + (size_t)(n0 + col16) * GG + HH; // enc part cols
        v8f acc = {};
        for (int k0 = 0; k0 < DD; k0 += 32) {
            union { v16h v; v8h h[2]; } a;
            a.v = *(const v16h*)(Arow + k0 + hi * 16);
            v16h bf = *(const v16h*)(Br + k0 + hi * 16);
            acc = __builtin_amdgcn_wmma_f32_16x16x32_f16(false, a.v, false, bf,
                                                         (short)0, acc, false, false);
        }
        const int col = n0 + col16;
        const float ph = preh[(size_t)b * HH + col];
        const float vc = vvec[col];
#pragma unroll
        for (int i = 0; i < 8; ++i)
            sacc[i] += tanhf(acc[i] + ph) * vc;
    }

    // Reduce partial sums over the 16 lanes of each half (cols -> row totals).
#pragma unroll
    for (int i = 0; i < 8; ++i) {
        float s = sacc[i];
        s += __shfl_xor(s, 1);
        s += __shfl_xor(s, 2);
        s += __shfl_xor(s, 4);
        s += __shfl_xor(s, 8);
        sacc[i] = s;
    }
    if (col16 == 0) {
        const int base = b * TT + t0 + w * 16 + hi * 8; // hi=0: rows 0-7, hi=1: rows 8-15
#pragma unroll
        for (int i = 0; i < 8; ++i) scores[base + i] = sacc[i];
    }
}

// ---------------------------------------------------------------------------
__global__ void attn_softmax_kernel(const float* __restrict__ scores,
                                    float* __restrict__ attn_out) // (B,T)
{
    __shared__ float red[256];
    const int b = blockIdx.x, t = threadIdx.x;
    const float s = scores[b * TT + t];
    red[t] = s; __syncthreads();
    for (int off = 128; off > 0; off >>= 1) {
        if (t < off) red[t] = fmaxf(red[t], red[t + off]);
        __syncthreads();
    }
    const float m = red[0]; __syncthreads();
    const float e = expf(s - m);
    red[t] = e; __syncthreads();
    for (int off = 128; off > 0; off >>= 1) {
        if (t < off) red[t] += red[t + off];
        __syncthreads();
    }
    attn_out[b * TT + t] = e / red[0];
}

__global__ void context_kernel(const float* __restrict__ enc,      // (T,B,D)
                               const float* __restrict__ attnw,    // (B,T)
                               float* __restrict__ context)        // (B,D)
{
    __shared__ float w[TT];
    const int b = blockIdx.y;
    const int d = blockIdx.x * 256 + threadIdx.x;
    for (int t = threadIdx.x; t < TT; t += 256) w[t] = attnw[b * TT + t];
    __syncthreads();
    float acc = 0.f;
    for (int t = 0; t < TT; ++t)
        acc += w[t] * enc[((size_t)t * BB + b) * DD + d];
    context[(size_t)b * DD + d] = acc;
}

// ---------------------------------------------------------------------------
__global__ void f32_to_f16_kernel(const float* __restrict__ src,
                                  _Float16* __restrict__ dst, size_t n)
{
    size_t i = (size_t)blockIdx.x * blockDim.x + threadIdx.x;
    if (i < n) dst[i] = (_Float16)src[i];
}

// dst[r*ldd + c] = c<ncols ? src[r*ncols+c] : 0   (zero K-padding)
__global__ void pad_convert_kernel(const float* __restrict__ src,
                                   _Float16* __restrict__ dst,
                                   int rows, int ncols, int ldd)
{
    size_t i = (size_t)blockIdx.x * blockDim.x + threadIdx.x;
    if (i >= (size_t)rows * ldd) return;
    const int r = (int)(i / ldd), c = (int)(i % ldd);
    dst[i] = (c < ncols) ? (_Float16)src[(size_t)r * ncols + c] : (_Float16)0.f;
}

// A[b,c] = emb_W[word_input[b], c] (f16), zero K-padding
__global__ void gather_emb_kernel(const int* __restrict__ idx,
                                  const float* __restrict__ embW,
                                  _Float16* __restrict__ A)
{
    const int b = blockIdx.y;
    const int c = blockIdx.x * 256 + threadIdx.x;
    if (c < KPAD)
        A[(size_t)b * KPAD + c] =
            (c < CC) ? (_Float16)embW[(size_t)idx[b] * CC + c] : (_Float16)0.f;
}

// x = [word_emb (H) | context (D)] -> f16
__global__ void build_x_kernel(const float* __restrict__ wemb,
                               const float* __restrict__ ctx,
                               _Float16* __restrict__ x16)
{
    const int b = blockIdx.y;
    const int j = blockIdx.x * 256 + threadIdx.x;
    const float v = (j < HH) ? wemb[(size_t)b * HH + j] : ctx[(size_t)b * DD + (j - HH)];
    x16[(size_t)b * GG + j] = (_Float16)v;
}

// GRU cell (PyTorch gate order r,z,n); gi/gh already include biases.
__global__ void gru_kernel(const float* __restrict__ gi, const float* __restrict__ gh,
                           const float* __restrict__ hprev,
                           float* __restrict__ hidden_out, _Float16* __restrict__ h16out)
{
    const int i = blockIdx.x * 256 + threadIdx.x;   // B*H
    const int b = i >> 9, h = i & 511;
    const size_t base = (size_t)b * GG;
    const float r = 1.f / (1.f + expf(-(gi[base + h]        + gh[base + h])));
    const float z = 1.f / (1.f + expf(-(gi[base + HH + h]   + gh[base + HH + h])));
    const float n = tanhf(gi[base + 2 * HH + h] + r * gh[base + 2 * HH + h]);
    const float hp = hprev[(size_t)b * HH + h];
    const float hn = (1.f - z) * n + z * hp;
    hidden_out[(size_t)b * HH + h] = hn;
    h16out[(size_t)b * HH + h] = (_Float16)hn;
}

__global__ void logsoftmax_kernel(const float* __restrict__ logits, // (B, CPAD)
                                  float* __restrict__ out)          // (B, C)
{
    __shared__ float red[256];
    const int b = blockIdx.x, tid = threadIdx.x;
    const float* row = logits + (size_t)b * CPAD;
    float m = -3.4e38f;
    for (int c = tid; c < CC; c += 256) m = fmaxf(m, row[c]);
    red[tid] = m; __syncthreads();
    for (int off = 128; off > 0; off >>= 1) {
        if (tid < off) red[tid] = fmaxf(red[tid], red[tid + off]);
        __syncthreads();
    }
    m = red[0]; __syncthreads();
    float s = 0.f;
    for (int c = tid; c < CC; c += 256) s += expf(row[c] - m);
    red[tid] = s; __syncthreads();
    for (int off = 128; off > 0; off >>= 1) {
        if (tid < off) red[tid] += red[tid + off];
        __syncthreads();
    }
    const float lse = m + logf(red[0]);
    for (int c = tid; c < CC; c += 256) out[(size_t)b * CC + c] = row[c] - lse;
}

// ---------------------------------------------------------------------------
extern "C" void kernel_launch(void* const* d_in, const int* in_sizes, int n_in,
                              void* d_out, int out_size, void* d_ws, size_t ws_size,
                              hipStream_t stream)
{
    (void)in_sizes; (void)n_in; (void)out_size; (void)ws_size;

    const int*   word_input = (const int*)  d_in[0];
    const float* last_hidden= (const float*)d_in[1];   // (1,B,H)
    const float* enc        = (const float*)d_in[2];   // (T,B,D)
    const float* emb_W      = (const float*)d_in[3];   // (C,C)
    const float* wl_W       = (const float*)d_in[4];   // (H,C)
    const float* wl_b       = (const float*)d_in[5];
    const float* attn_W     = (const float*)d_in[6];   // (H,G)
    const float* attn_b     = (const float*)d_in[7];
    const float* vvec       = (const float*)d_in[8];
    const float* gru_Wih    = (const float*)d_in[9];   // (G,G)
    const float* gru_Whh    = (const float*)d_in[10];  // (G,H)
    const float* gru_bih    = (const float*)d_in[11];
    const float* gru_bhh    = (const float*)d_in[12];
    const float* out_W      = (const float*)d_in[13];  // (C,H)
    const float* out_b      = (const float*)d_in[14];

    float* out_logsm  = (float*)d_out;                       // (B,C)
    float* out_hidden = out_logsm + (size_t)BB * CC;         // (B,H)
    float* out_attn   = out_hidden + (size_t)BB * HH;        // (B,T)

    // Workspace bump allocator (256-byte aligned).
    char* ws = (char*)d_ws;
    size_t off = 0;
    auto alloc = [&](size_t bytes) {
        char* p = ws + off;
        off += (bytes + 255) & ~(size_t)255;
        return p;
    };
    _Float16* attnW16  = (_Float16*)alloc((size_t)HH * GG * 2);
    _Float16* Wih16    = (_Float16*)alloc((size_t)GG * GG * 2);
    _Float16* Whh16    = (_Float16*)alloc((size_t)GG * HH * 2);
    _Float16* outW16   = (_Float16*)alloc((size_t)CC * HH * 2);
    _Float16* wlW16    = (_Float16*)alloc((size_t)HH * KPAD * 2);
    _Float16* onehot16 = (_Float16*)alloc((size_t)BB * KPAD * 2);
    _Float16* h16      = (_Float16*)alloc((size_t)BB * HH * 2);
    _Float16* x16      = (_Float16*)alloc((size_t)BB * GG * 2);
    _Float16* hid16    = (_Float16*)alloc((size_t)BB * HH * 2);
    float* preh    = (float*)alloc((size_t)BB * HH * 4);
    float* wemb    = (float*)alloc((size_t)BB * HH * 4);
    float* scores  = (float*)alloc((size_t)BB * TT * 4);
    float* ctx     = (float*)alloc((size_t)BB * DD * 4);
    float* gi      = (float*)alloc((size_t)BB * GG * 4);
    float* gh      = (float*)alloc((size_t)BB * GG * 4);
    float* logits  = (float*)alloc((size_t)BB * CPAD * 4);

    const int TPB = 256;
    auto blk = [](size_t n) { return (unsigned)((n + 255) / 256); };

    // --- weight / activation conversions ---------------------------------
    f32_to_f16_kernel<<<blk((size_t)HH * GG), TPB, 0, stream>>>(attn_W, attnW16, (size_t)HH * GG);
    f32_to_f16_kernel<<<blk((size_t)GG * GG), TPB, 0, stream>>>(gru_Wih, Wih16, (size_t)GG * GG);
    f32_to_f16_kernel<<<blk((size_t)GG * HH), TPB, 0, stream>>>(gru_Whh, Whh16, (size_t)GG * HH);
    f32_to_f16_kernel<<<blk((size_t)CC * HH), TPB, 0, stream>>>(out_W, outW16, (size_t)CC * HH);
    f32_to_f16_kernel<<<blk((size_t)BB * HH), TPB, 0, stream>>>(last_hidden, h16, (size_t)BB * HH);
    pad_convert_kernel<<<blk((size_t)HH * KPAD), TPB, 0, stream>>>(wl_W, wlW16, HH, CC, KPAD);
    gather_emb_kernel<<<dim3((KPAD + 255) / 256, BB), TPB, 0, stream>>>(word_input, emb_W, onehot16);

    // --- small GEMMs (WMMA) ----------------------------------------------
    // pre_h = h @ attn_W[:, :H].T + attn_b     (B,H)
    wmma_gemm_kernel<<<dim3(HH / 16, BB / 16), 32, 0, stream>>>(
        h16, HH, attnW16, GG, HH, attn_b, preh, HH, HH);
    // gh = h @ Whh.T + bhh                      (B,3H)
    wmma_gemm_kernel<<<dim3(GG / 16, BB / 16), 32, 0, stream>>>(
        h16, HH, Whh16, HH, GG, gru_bhh, gh, GG, HH);
    // word_emb = emb_W[idx] @ wl_W.T + wl_b     (B,H), K = C padded
    wmma_gemm_kernel<<<dim3(HH / 16, BB / 16), 32, 0, stream>>>(
        onehot16, KPAD, wlW16, KPAD, HH, wl_b, wemb, HH, KPAD);

    // --- fused attention scores (big WMMA GEMM + tanh + v-dot) -----------
    attn_scores_kernel<<<dim3(TT / 64, BB), 128, 64 * DD * sizeof(_Float16), stream>>>(
        enc, attnW16, preh, vvec, scores);
    attn_softmax_kernel<<<BB, TT, 0, stream>>>(scores, out_attn);
    context_kernel<<<dim3(DD / 256, BB), TPB, 0, stream>>>(enc, out_attn, ctx);

    // --- GRU --------------------------------------------------------------
    build_x_kernel<<<dim3(GG / 256, BB), TPB, 0, stream>>>(wemb, ctx, x16);
    wmma_gemm_kernel<<<dim3(GG / 16, BB / 16), 32, 0, stream>>>(
        x16, GG, Wih16, GG, GG, gru_bih, gi, GG, GG);
    gru_kernel<<<blk((size_t)BB * HH), TPB, 0, stream>>>(gi, gh, last_hidden, out_hidden, hid16);

    // --- output projection + log_softmax ----------------------------------
    wmma_gemm_kernel<<<dim3(CPAD / 16, BB / 16), 32, 0, stream>>>(
        hid16, HH, outW16, HH, CC, out_b, logits, CPAD, HH);
    logsoftmax_kernel<<<BB, TPB, 0, stream>>>(logits, out_logsm);
}
